// GNN_63084479643907
// MI455X (gfx1250) — compile-verified, matrix operand order checked by reading, not run on hardware
//
#include <hip/hip_runtime.h>
#include <hip/hip_bf16.h>

// ---------------------------------------------------------------------------
// HGT forward for MI455X (gfx1250).
// - Dense per-type linears use v_wmma_f32_16x16x32_bf16: each 16-node tile is
//   multiplied against all 4 type weights (4 accumulators/wave), output row
//   selects by node_type. 4x FLOP replication is free vs the WMMA ceiling.
// - Weights repacked once per launch into fragment-ordered bf16: each lane
//   loads its whole 16-elem B fragment as one 32B vector (2x b128).
// - A tile staged to LDS as bf16 pairs: 1x global_load_b64 + 1x
//   v_cvt_pk_bf16_f32 + 1x ds_store_b32 per thread per K-step.
// - gelu fused into a-GEMM input staging; sigmoid-skip blend fused into its
//   store -> no standalone elementwise passes over HBM.
// - Edge softmax/aggregation = gather + atomic kernels (bandwidth bound).
// ---------------------------------------------------------------------------

#define N_TYPES 4
#define N_REL   8
#define N_HEADS 8
#define IN_DIM  256
#define N_HID   128
#define D_K     16
#define N_LAYERS 2
#define NNODES  100000
#define NEDGES  400000

typedef __attribute__((ext_vector_type(16))) __bf16 v16bf;
typedef __attribute__((ext_vector_type(8)))  float  v8f;

__device__ __forceinline__ unsigned fflip(float f) {
    unsigned u = __float_as_uint(f);
    return (u & 0x80000000u) ? ~u : (u | 0x80000000u);
}
__device__ __forceinline__ float funflip(unsigned u) {
    return __uint_as_float((u & 0x80000000u) ? (u & 0x7FFFFFFFu) : ~u);
}
__device__ __forceinline__ float gelu_f(float v) {
    float c = 0.7978845608028654f * (v + 0.044715f * v * v * v);
    return 0.5f * v * (1.0f + tanhf(c));
}

// ---------------------------------------------------------------------------
// Repack w[T, KDIM, N_HID] (f32) -> fw[T][KDIM/32][8 ntile][32 lane][16] bf16
// fragment order matching the ISA §7.12.2 16-bit B layout.
// ---------------------------------------------------------------------------
template <int KDIM>
__global__ __launch_bounds__(256) void repack_weights(
    const float* __restrict__ w, __bf16* __restrict__ fw)
{
    const int total = N_TYPES * (KDIM / 32) * 8 * 32 * 16;
    int gid = blockIdx.x * blockDim.x + threadIdx.x;
    if (gid >= total) return;
    int e     = gid & 15;
    int lane  = (gid >> 4) & 31;
    int ntile = (gid >> 9) & 7;
    int rest  = gid >> 12;               // t*(KDIM/32) + ks
    int ks    = rest % (KDIM / 32);
    int t     = rest / (KDIM / 32);
    int kblk  = lane >> 4, col = lane & 15;
    int kk    = (e < 8) ? (kblk * 8 + e) : (16 + kblk * 8 + (e - 8));
    int k     = ks * 32 + kk;
    int n     = ntile * 16 + col;
    fw[gid] = (__bf16)w[((size_t)t * KDIM + k) * N_HID + n];
}

// ---------------------------------------------------------------------------
// Per-node-type linear with optional fused gelu-on-input, tanh-on-output,
// and sigmoid-skip blend on store.
// Block = 256 threads = 8 waves; block owns 16 node rows, wave w owns output
// cols [16w, 16w+16).
// ---------------------------------------------------------------------------
template <int KDIM, bool GELU_IN, bool TANH_OUT, bool BLEND>
__global__ __launch_bounds__(256) void pertype_gemm(
    const float*  __restrict__ x,      // [M, KDIM]
    const int*    __restrict__ ntype,  // [M]
    const __bf16* __restrict__ fw,     // fragment-packed weights
    const float*  __restrict__ b,      // [N_TYPES, N_HID]
    const float*  __restrict__ hin,    // [M, N_HID] (BLEND only)
    const float*  __restrict__ skipv,  // [N_TYPES]  (BLEND only)
    float* __restrict__ out,           // [M, N_HID]
    int M)
{
    constexpr int KSTEPS = KDIM / 32;
    __shared__ __bf16 sA[16][32];
    __shared__ int    sType[16];

    const int m0   = blockIdx.x * 16;
    const int wave = threadIdx.x >> 5;
    const int lane = threadIdx.x & 31;
    const int n0   = wave * 16;
    const int rowT = lane & 15;
    const int col  = lane & 15;
    const int kblk = lane >> 4;

    // staging coords: thread -> one bf16 pair of the 16x32 tile
    const int sRow  = threadIdx.x >> 4;       // 0..15
    const int sPair = threadIdx.x & 15;       // 0..15 (2 cols each)
    const int sGr   = m0 + sRow;
    const float* xRow = x + (size_t)sGr * KDIM + 2 * sPair;

    v8f acc[N_TYPES] = {};

    if (threadIdx.x < 16) {
        int r = m0 + threadIdx.x;
        sType[threadIdx.x] = (r < M) ? ntype[r] : 0;
    }

    const v16bf* fwv = (const v16bf*)fw;

    for (int ks = 0; ks < KSTEPS; ++ks) {
        __syncthreads();
        {
            float v0 = 0.0f, v1 = 0.0f;
            if (sGr < M) {
                v0 = xRow[ks * 32 + 0];
                v1 = xRow[ks * 32 + 1];
            }
            if (GELU_IN) { v0 = gelu_f(v0); v1 = gelu_f(v1); }
            union { __bf16 h[2]; unsigned u; } pk;
            pk.h[0] = (__bf16)v0;
            pk.h[1] = (__bf16)v1;
            ((unsigned*)sA)[threadIdx.x] = pk.u;  // = sA[sRow][2*sPair..+1]
        }
        __syncthreads();

        v16bf afrag;
#pragma unroll
        for (int e = 0; e < 16; ++e) {
            int kk = (e < 8) ? (kblk * 8 + e) : (16 + kblk * 8 + (e - 8));
            afrag[e] = sA[rowT][kk];
        }

#pragma unroll
        for (int t = 0; t < N_TYPES; ++t) {
            v16bf bfrag = fwv[(((size_t)t * KSTEPS + ks) * 8 + wave) * 32 + lane];
            acc[t] = __builtin_amdgcn_wmma_f32_16x16x32_bf16(
                false, afrag, false, bfrag, (short)0, acc[t], false, false);
        }
    }
    __syncthreads();

#pragma unroll
    for (int i = 0; i < 8; ++i) {
        int r  = i + 8 * kblk;
        int gr = m0 + r;
        if (gr >= M) continue;
        int t = sType[r];
        float cv = (t == 0) ? acc[0][i]
                 : (t == 1) ? acc[1][i]
                 : (t == 2) ? acc[2][i]
                 :            acc[3][i];
        float val = cv + b[t * N_HID + n0 + col];
        if (TANH_OUT) val = tanhf(val);
        size_t oidx = (size_t)gr * N_HID + n0 + col;
        if (BLEND) {
            float a = 1.0f / (1.0f + expf(-skipv[t]));
            val = val * a + hin[oidx] * (1.0f - a);
        }
        out[oidx] = val;
    }
}

// ---------------------------------------------------------------------------
// Edge pass 1: score + segment-max (flipped-uint atomicMax)
// ---------------------------------------------------------------------------
__global__ __launch_bounds__(256) void edge_score_max(
    const int* __restrict__ src, const int* __restrict__ tgt,
    const int* __restrict__ etype,
    const float* __restrict__ k, const float* __restrict__ q,
    const float* __restrict__ attm, const float* __restrict__ pri,
    float* __restrict__ score, unsigned* __restrict__ mflip, int E)
{
    int gid = blockIdx.x * blockDim.x + threadIdx.x;
    if (gid >= E * N_HEADS) return;
    int e = gid >> 3, h = gid & 7;
    int s = src[e], t = tgt[e], r = etype[e];
    const float* kv = k + (size_t)s * N_HID + h * D_K;
    const float* qv = q + (size_t)t * N_HID + h * D_K;
    const float* A  = attm + ((size_t)r * N_HEADS + h) * D_K * D_K;

    float sc = 0.0f;
#pragma unroll
    for (int f = 0; f < D_K; ++f) {
        float kr = 0.0f;
#pragma unroll
        for (int d = 0; d < D_K; ++d) kr += kv[d] * A[d * D_K + f];
        sc += qv[f] * kr;
    }
    sc *= pri[r * N_HEADS + h] * 0.25f;  // 1/sqrt(16)
    score[gid] = sc;
    atomicMax(mflip + (size_t)t * N_HEADS + h, fflip(sc));
}

// Edge pass 2: z[tgt,h] += exp(score - m)
__global__ __launch_bounds__(256) void edge_expsum(
    const int* __restrict__ tgt, const float* __restrict__ score,
    const unsigned* __restrict__ mflip, float* __restrict__ z, int E)
{
    int gid = blockIdx.x * blockDim.x + threadIdx.x;
    if (gid >= E * N_HEADS) return;
    int e = gid >> 3, h = gid & 7;
    size_t th = (size_t)tgt[e] * N_HEADS + h;
    atomicAdd(&z[th], expf(score[gid] - funflip(mflip[th])));
}

// Edge pass 3: agg[tgt,h,:] += att * (v_src @ msg_m[r,h])
__global__ __launch_bounds__(256) void edge_aggregate(
    const int* __restrict__ src, const int* __restrict__ tgt,
    const int* __restrict__ etype,
    const float* __restrict__ v, const float* __restrict__ msgm,
    const float* __restrict__ score, const unsigned* __restrict__ mflip,
    const float* __restrict__ z, float* __restrict__ agg, int E)
{
    int gid = blockIdx.x * blockDim.x + threadIdx.x;
    if (gid >= E * N_HEADS) return;
    int e = gid >> 3, h = gid & 7;
    int s = src[e], t = tgt[e], r = etype[e];
    size_t th = (size_t)t * N_HEADS + h;
    float att = expf(score[gid] - funflip(mflip[th])) / (z[th] + 1e-9f);
    const float* vv = v + (size_t)s * N_HID + h * D_K;
    const float* Mm = msgm + ((size_t)r * N_HEADS + h) * D_K * D_K;
    float* ag = agg + (size_t)t * N_HID + h * D_K;
#pragma unroll
    for (int f = 0; f < D_K; ++f) {
        float vr = 0.0f;
#pragma unroll
        for (int d = 0; d < D_K; ++d) vr += vv[d] * Mm[d * D_K + f];
        atomicAdd(ag + f, att * vr);
    }
}

extern "C" void kernel_launch(void* const* d_in, const int* in_sizes, int n_in,
                              void* d_out, int out_size, void* d_ws, size_t ws_size,
                              hipStream_t stream)
{
    const float* x        = (const float*)d_in[0];
    const int*   ntype    = (const int*)  d_in[1];
    const int*   eidx     = (const int*)  d_in[2];
    const int*   etype    = (const int*)  d_in[3];
    const float* adapt_w  = (const float*)d_in[4];
    const float* adapt_b  = (const float*)d_in[5];
    const float* k_w      = (const float*)d_in[6];
    const float* k_b      = (const float*)d_in[7];
    const float* q_w      = (const float*)d_in[8];
    const float* q_b      = (const float*)d_in[9];
    const float* v_w      = (const float*)d_in[10];
    const float* v_b      = (const float*)d_in[11];
    const float* a_w      = (const float*)d_in[12];
    const float* a_b      = (const float*)d_in[13];
    const float* rel_pri  = (const float*)d_in[14];
    const float* rel_att  = (const float*)d_in[15];
    const float* rel_msg  = (const float*)d_in[16];
    const float* skip     = (const float*)d_in[17];

    const int* src = eidx;
    const int* tgt = eidx + NEDGES;

    char*  ws  = (char*)d_ws;
    size_t off = 0;
    auto carve = [&](size_t bytes) -> void* {
        void* p = ws + off;
        off += (bytes + 255) & ~(size_t)255;
        return p;
    };
    const size_t nh_bytes = (size_t)NNODES * N_HID * sizeof(float);
    float*    hA    = (float*)   carve(nh_bytes);
    float*    hB    = (float*)   carve(nh_bytes);
    float*    kbuf  = (float*)   carve(nh_bytes);   // reused as agg after pass 1
    float*    qbuf  = (float*)   carve(nh_bytes);
    float*    vbuf  = (float*)   carve(nh_bytes);
    float*    score = (float*)   carve((size_t)NEDGES * N_HEADS * sizeof(float));
    unsigned* mflip = (unsigned*)carve((size_t)NNODES * N_HEADS * sizeof(unsigned));
    float*    zbuf  = (float*)   carve((size_t)NNODES * N_HEADS * sizeof(float));

    // fragment-packed bf16 weights
    const int FRAG256 = N_TYPES * (IN_DIM / 32) * 8 * 32 * 16;  // 131072
    const int FRAG128 = N_TYPES * (N_HID  / 32) * 8 * 32 * 16;  // 65536
    __bf16* fw_adapt = (__bf16*)carve((size_t)FRAG256 * sizeof(__bf16));
    __bf16* fw_k[2], *fw_q[2], *fw_v[2], *fw_a[2];
    for (int l = 0; l < N_LAYERS; ++l) {
        fw_k[l] = (__bf16*)carve((size_t)FRAG128 * sizeof(__bf16));
        fw_q[l] = (__bf16*)carve((size_t)FRAG128 * sizeof(__bf16));
        fw_v[l] = (__bf16*)carve((size_t)FRAG128 * sizeof(__bf16));
        fw_a[l] = (__bf16*)carve((size_t)FRAG128 * sizeof(__bf16));
    }
    (void)ws_size; (void)in_sizes; (void)n_in; (void)out_size;

    repack_weights<IN_DIM><<<FRAG256 / 256, 256, 0, stream>>>(adapt_w, fw_adapt);
    for (int l = 0; l < N_LAYERS; ++l) {
        const size_t wOff = (size_t)l * N_TYPES * N_HID * N_HID;
        repack_weights<N_HID><<<FRAG128 / 256, 256, 0, stream>>>(k_w + wOff, fw_k[l]);
        repack_weights<N_HID><<<FRAG128 / 256, 256, 0, stream>>>(q_w + wOff, fw_q[l]);
        repack_weights<N_HID><<<FRAG128 / 256, 256, 0, stream>>>(v_w + wOff, fw_v[l]);
        repack_weights<N_HID><<<FRAG128 / 256, 256, 0, stream>>>(a_w + wOff, fw_a[l]);
    }

    const int nodeTiles = (NNODES + 15) / 16;              // 6250
    const int edgeBlks  = (NEDGES * N_HEADS + 255) / 256;  // 12500

    // adapter: h = tanh(per-type linear 256->128)
    pertype_gemm<IN_DIM, false, true, false><<<nodeTiles, 256, 0, stream>>>(
        x, ntype, fw_adapt, adapt_b, nullptr, nullptr, hA, NNODES);

    for (int l = 0; l < N_LAYERS; ++l) {
        const float* h_in  = (l == 0) ? hA : hB;
        float*       h_out = (l == 0) ? hB : (float*)d_out;

        const size_t bOff = (size_t)l * N_TYPES * N_HID;
        const size_t rOff = (size_t)l * N_REL * N_HEADS * D_K * D_K;

        pertype_gemm<N_HID, false, false, false><<<nodeTiles, 256, 0, stream>>>(
            h_in, ntype, fw_k[l], k_b + bOff, nullptr, nullptr, kbuf, NNODES);
        pertype_gemm<N_HID, false, false, false><<<nodeTiles, 256, 0, stream>>>(
            h_in, ntype, fw_q[l], q_b + bOff, nullptr, nullptr, qbuf, NNODES);
        pertype_gemm<N_HID, false, false, false><<<nodeTiles, 256, 0, stream>>>(
            h_in, ntype, fw_v[l], v_b + bOff, nullptr, nullptr, vbuf, NNODES);

        hipMemsetAsync(mflip, 0, (size_t)NNODES * N_HEADS * sizeof(unsigned), stream);

        edge_score_max<<<edgeBlks, 256, 0, stream>>>(
            src, tgt, etype, kbuf, qbuf,
            rel_att + rOff, rel_pri + (size_t)l * N_REL * N_HEADS,
            score, mflip, NEDGES);

        float* agg = kbuf;  // kbuf dead after pass 1
        hipMemsetAsync(zbuf, 0, (size_t)NNODES * N_HEADS * sizeof(float), stream);
        hipMemsetAsync(agg, 0, nh_bytes, stream);

        edge_expsum<<<edgeBlks, 256, 0, stream>>>(tgt, score, mflip, zbuf, NEDGES);

        edge_aggregate<<<edgeBlks, 256, 0, stream>>>(
            src, tgt, etype, vbuf, rel_msg + rOff, score, mflip, zbuf, agg, NEDGES);

        // fused: trans = per-type-linear(gelu(agg)); h_out = blend(trans, h_in)
        pertype_gemm<N_HID, true, false, true><<<nodeTiles, 256, 0, stream>>>(
            agg, ntype, fw_a[l], a_b + bOff, h_in, skip + (size_t)l * N_TYPES,
            h_out, NNODES);
    }
}
